// MambaBlocks_48120813584486
// MI455X (gfx1250) — matrix operand order repaired and checked by reading, not compiled, not run on hardware
//
#include <hip/hip_runtime.h>
#include <hip/hip_bf16.h>

// ---- problem constants (from reference) ----
#define LAYERS 4
#define DM     512     // D_MODEL
#define DI     1024    // D_INNER
#define DS2    64      // D_STATE
#define DCONV  4
#define DTR    32      // DT_RANK
#define BN     8
#define TN     2048
#define XPF    (DTR + 2 * DS2)   // 160

static const size_t BT = (size_t)BN * TN;   // 16384

typedef __attribute__((ext_vector_type(16))) _Float16 v16h;
typedef __attribute__((ext_vector_type(8)))  _Float16 v8h;
typedef __attribute__((ext_vector_type(8)))  float    v8f;

// ---------------------------------------------------------------------------
// WMMA C/D fragment helpers (ISA 7.12.2 layout, wave32)
// 32-bit C/D 16x16: VGPR v -> row v (lanes 0-15, col=lane) / row v+8 (16-31)
// ---------------------------------------------------------------------------
__device__ __forceinline__ v8f load_c_frag(const float* __restrict__ C,
                                           int ldc, int row0, int col0, int lane) {
  int n  = lane & 15;
  int mo = (lane & 16) ? 8 : 0;
  v8f c;
#pragma unroll
  for (int v = 0; v < 8; ++v)
    c[v] = C[(size_t)(row0 + mo + v) * ldc + col0 + n];
  return c;
}

__device__ __forceinline__ void store_c_frag(float* __restrict__ C, int ldc,
                                             int row0, int col0, int lane, v8f c) {
  int n  = lane & 15;
  int mo = (lane & 16) ? 8 : 0;
#pragma unroll
  for (int v = 0; v < 8; ++v)
    C[(size_t)(row0 + mo + v) * ldc + col0 + n] = c[v];
}

// ---------------------------------------------------------------------------
// GEMM:  C[M,N] (f32) = [ACC: C + ] A[M,K] (f16, row-major) x W[N,K]^T (f16)
//
// Each wave computes a 16x64 strip of C (4 N-tiles): one A fragment feeds
// 4 v_wmma_f32_16x16x32_f16 per K step (4x A reuse).  Block = 8 waves
// arranged 2 (M) x 4 (N) -> 32x256 block tile.
//
// FULL=true  : N % 256 == 0, no tile guards -> straight-line inner loop.
// FULL=false : ragged N, guards derived from readfirstlane'd wave index so
//              they stay scalar branches (EXEC all-ones across WMMA).
//
// Fragment layouts (ISA 7.12.2):
//  A 16x32 f16: lanes 0-15 -> M=lane, K 0..7/16..23; lanes 16-31 -> K 8..15/24..31
//  B 32x16 f16 (=W^T): lanes 0-15 -> N=lane, K 0..7/16..23; lanes 16-31 -> K 8..15/24..31
// Loop-invariant base pointers are hoisted; the K loop indexes with +k0 only.
// ---------------------------------------------------------------------------
template <bool ACC, bool FULL>
__global__ __launch_bounds__(256) void gemm_wmma_f16(
    const _Float16* __restrict__ A, const _Float16* __restrict__ W,
    float* __restrict__ C, int M, int N, int K) {
  const int lane = threadIdx.x & 31;
  const int wave = __builtin_amdgcn_readfirstlane(threadIdx.x >> 5);  // uniform
  const int row0    = blockIdx.y * 32 + (wave >> 2) * 16;
  const int colbase = blockIdx.x * 256 + (wave & 3) * 64;
  if (row0 >= M) return;                       // uniform
  if (!FULL && colbase >= N) return;           // uniform

  const int  m  = lane & 15;
  const bool hi = (lane & 16) != 0;

  // hoisted fragment base pointers (K-loop adds only +k0)
  const _Float16* pA = A + (size_t)(row0 + m) * K + (hi ? 8 : 0);
  const _Float16* pW[4];
  bool valid[4];
#pragma unroll
  for (int j = 0; j < 4; ++j) {
    valid[j] = FULL || ((colbase + j * 16) < N);     // scalar condition
    const int col = valid[j] ? (colbase + j * 16) : 0;
    pW[j] = W + (size_t)(col + m) * K + (hi ? 8 : 0);
  }

  v8f c[4];
#pragma unroll
  for (int j = 0; j < 4; ++j) {
    if (ACC && valid[j]) {
      c[j] = load_c_frag(C, N, row0, colbase + j * 16, lane);
    } else {
#pragma unroll
      for (int v = 0; v < 8; ++v) c[j][v] = 0.0f;
    }
  }

  for (int k0 = 0; k0 < K; k0 += 32) {
    v8h alo = *reinterpret_cast<const v8h*>(pA + k0);
    v8h ahi = *reinterpret_cast<const v8h*>(pA + k0 + 16);
    v16h a;
#pragma unroll
    for (int i = 0; i < 8; ++i) { a[i] = alo[i]; a[8 + i] = ahi[i]; }

    if (k0 + 32 < K) {
      // pull next K-step lines toward the WGP (global_prefetch_b8)
      __builtin_prefetch(pA + k0 + 32, 0, 3);
      __builtin_prefetch(pW[0] + k0 + 32, 0, 3);
    }

#pragma unroll
    for (int j = 0; j < 4; ++j) {
      if (valid[j]) {
        v8h b0 = *reinterpret_cast<const v8h*>(pW[j] + k0);
        v8h b1 = *reinterpret_cast<const v8h*>(pW[j] + k0 + 16);
        v16h b;
#pragma unroll
        for (int i = 0; i < 8; ++i) { b[i] = b0[i]; b[8 + i] = b1[i]; }
        c[j] = __builtin_amdgcn_wmma_f32_16x16x32_f16(
            false, a, false, b, (short)0, c[j], false, false);
      }
    }
  }

#pragma unroll
  for (int j = 0; j < 4; ++j)
    if (valid[j]) store_c_frag(C, N, row0, colbase + j * 16, lane, c[j]);
}

// ---------------------------------------------------------------------------
// Elementwise / data-movement kernels
// ---------------------------------------------------------------------------
__global__ void k_f32_to_f16(const float* __restrict__ src,
                             _Float16* __restrict__ dst, size_t n) {
  size_t i = (size_t)blockIdx.x * blockDim.x + threadIdx.x;
  if (i < n) dst[i] = (_Float16)src[i];
}

// x: (B, DM, T) -> h: (B, T, DM)
__global__ void k_transpose_in(const float* __restrict__ x, float* __restrict__ h) {
  size_t i = (size_t)blockIdx.x * blockDim.x + threadIdx.x;
  size_t total = BT * DM;
  if (i >= total) return;
  int d = (int)(i % DM);
  size_t bt = i / DM;
  int t = (int)(bt % TN);
  int b = (int)(bt / TN);
  h[i] = x[((size_t)b * DM + d) * TN + t];
}

// h: (B, T, DM) -> out: (B, DM, T)
__global__ void k_transpose_out(const float* __restrict__ h, float* __restrict__ out) {
  size_t i = (size_t)blockIdx.x * blockDim.x + threadIdx.x;
  size_t total = BT * DM;
  if (i >= total) return;
  int d = (int)(i % DM);
  size_t bt = i / DM;
  int t = (int)(bt % TN);
  int b = (int)(bt / TN);
  out[((size_t)b * DM + d) * TN + t] = h[i];
}

// LayerNorm over DM per token; outputs f16 activations for the GEMM.
__global__ __launch_bounds__(256) void k_layernorm(
    const float* __restrict__ X, const float* __restrict__ w,
    const float* __restrict__ bias, _Float16* __restrict__ Y) {
  const size_t tok = blockIdx.x;
  const float* x = X + tok * DM;
  float s = 0.f, s2 = 0.f;
  for (int i = threadIdx.x; i < DM; i += 256) {
    float v = x[i];
    s += v; s2 += v * v;
  }
#pragma unroll
  for (int off = 16; off > 0; off >>= 1) {
    s  += __shfl_xor(s,  off, 32);
    s2 += __shfl_xor(s2, off, 32);
  }
  __shared__ float rs[8], rs2[8];
  int lane = threadIdx.x & 31, wave = threadIdx.x >> 5;
  if (lane == 0) { rs[wave] = s; rs2[wave] = s2; }
  __syncthreads();
  float S = 0.f, S2 = 0.f;
#pragma unroll
  for (int i = 0; i < 8; ++i) { S += rs[i]; S2 += rs2[i]; }
  float mu  = S / DM;
  float var = S2 / DM - mu * mu;
  float rsd = rsqrtf(var + 1e-5f);
  _Float16* y = Y + tok * DM;
  for (int i = threadIdx.x; i < DM; i += 256)
    y[i] = (_Float16)((x[i] - mu) * rsd * w[i] + bias[i]);
}

// Causal depthwise conv (width 4) + SiLU over xi = xz[:, :DI].
// Writes xc (f32, for the scan) and xc16 (f16, for x_proj GEMM).
__global__ void k_conv_silu(const float* __restrict__ xz,
                            const float* __restrict__ cw,
                            const float* __restrict__ cb,
                            float* __restrict__ xc, _Float16* __restrict__ xc16) {
  size_t i = (size_t)blockIdx.x * blockDim.x + threadIdx.x;
  size_t total = BT * DI;
  if (i >= total) return;
  int d = (int)(i % DI);
  size_t bt = i / DI;
  int t = (int)(bt % TN);
  size_t b = bt / TN;
  float acc = cb[d];
#pragma unroll
  for (int j = 0; j < DCONV; ++j) {
    int tt = t - (DCONV - 1) + j;
    if (tt >= 0)
      acc += cw[d * DCONV + j] * xz[(b * TN + tt) * (2 * DI) + d];
  }
  float sv = acc / (1.0f + __expf(-acc));   // SiLU
  xc[i]   = sv;
  xc16[i] = (_Float16)sv;
}

// dt slice of x_dbl -> f16 for the dt_proj GEMM
__global__ void k_dt_convert(const float* __restrict__ xdbl, _Float16* __restrict__ dt16) {
  size_t i = (size_t)blockIdx.x * blockDim.x + threadIdx.x;
  size_t total = BT * DTR;
  if (i >= total) return;
  int r = (int)(i % DTR);
  size_t bt = i / DTR;
  dt16[i] = (_Float16)xdbl[bt * XPF + r];
}

// delta = softplus(delta_raw + dtb)
__global__ void k_softplus_bias(float* __restrict__ delta, const float* __restrict__ dtb) {
  size_t i = (size_t)blockIdx.x * blockDim.x + threadIdx.x;
  size_t total = BT * DI;
  if (i >= total) return;
  int d = (int)(i % DI);
  float v = delta[i] + dtb[d];
  delta[i] = (v > 20.0f) ? v : log1pf(__expf(v));
}

// ---------------------------------------------------------------------------
// Selective scan: one thread per (b, d) channel, 64-state h kept in VGPRs,
// A row (-exp(A_log)) kept in VGPRs, B_t/C_t staged in LDS per timestep.
// Software-pipelined: step t+1's global loads (B,C,delta,x,z) issue before
// step t's 64-exp compute chain so VMEM latency hides under the TRANS ops.
// Fuses +x*D and the SiLU(z) gate; emits f16 for the out_proj GEMM.
// Block = 64 threads (64 channels of one batch), grid = B * (DI/64).
// ---------------------------------------------------------------------------
__global__ __launch_bounds__(64) void k_scan(
    const float* __restrict__ delta, const float* __restrict__ xc,
    const float* __restrict__ xdbl,  const float* __restrict__ xz,
    const float* __restrict__ A_log, const float* __restrict__ Dp,
    _Float16* __restrict__ g16) {
  const int b    = blockIdx.x >> 4;          // DI/64 == 16 blocks per batch
  const int dblk = blockIdx.x & 15;
  const int tid  = threadIdx.x;
  const int d    = dblk * 64 + tid;

  __shared__ float LB[DS2];
  __shared__ float LC[DS2];

  float Areg[DS2];
  float hst[DS2];
  const float* al = A_log + (size_t)d * DS2;
#pragma unroll
  for (int s = 0; s < DS2; ++s) {
    Areg[s] = -__expf(al[s]);
    hst[s]  = 0.0f;
  }
  const float Dpd = Dp[d];

  // prologue: loads for t = 0
  size_t bt0 = (size_t)b * TN;
  float nb = xdbl[bt0 * XPF + DTR + tid];
  float nc = xdbl[bt0 * XPF + DTR + DS2 + tid];
  float nd = delta[bt0 * DI + d];
  float nx = xc[bt0 * DI + d];
  float nz = xz[bt0 * (2 * DI) + DI + d];

  for (int t = 0; t < TN; ++t) {
    size_t bt = (size_t)b * TN + t;
    float dlt = nd, xv = nx, zv = nz;

    __syncthreads();                  // previous step done reading LB/LC
    LB[tid] = nb;
    LC[tid] = nc;
    __syncthreads();

    if (t + 1 < TN) {                 // issue next step's loads early
      size_t bt1 = bt + 1;
      nb = xdbl[bt1 * XPF + DTR + tid];
      nc = xdbl[bt1 * XPF + DTR + DS2 + tid];
      nd = delta[bt1 * DI + d];
      nx = xc[bt1 * DI + d];
      nz = xz[bt1 * (2 * DI) + DI + d];
    }

    float dx = dlt * xv;
    float y  = 0.0f;
#pragma unroll
    for (int s = 0; s < DS2; ++s) {
      float dA = __expf(dlt * Areg[s]);
      hst[s] = dA * hst[s] + dx * LB[s];
      y += hst[s] * LC[s];
    }
    float yss  = y + xv * Dpd;
    float gate = zv / (1.0f + __expf(-zv));  // SiLU(z)
    g16[bt * DI + d] = (_Float16)(yss * gate);
  }
}

// ---------------------------------------------------------------------------
// Host launcher
// ---------------------------------------------------------------------------
extern "C" void kernel_launch(void* const* d_in, const int* in_sizes, int n_in,
                              void* d_out, int out_size, void* d_ws, size_t ws_size,
                              hipStream_t stream) {
  const float* x      = (const float*)d_in[0];
  const float* ln_w   = (const float*)d_in[1];
  const float* ln_b   = (const float*)d_in[2];
  const float* in_w   = (const float*)d_in[3];
  const float* conv_w = (const float*)d_in[4];
  const float* conv_b = (const float*)d_in[5];
  const float* xp_w   = (const float*)d_in[6];
  const float* dt_w   = (const float*)d_in[7];
  const float* dt_b   = (const float*)d_in[8];
  const float* A_log  = (const float*)d_in[9];
  const float* Dp     = (const float*)d_in[10];
  const float* out_w  = (const float*)d_in[11];

  // ---- workspace layout (bytes, all 256B aligned) ----
  char* ws = (char*)d_ws;
  const size_t SZ_H     = BT * DM * 4;            // f32 residual stream
  const size_t SZ_Y16   = BT * DM * 2;
  const size_t SZ_XZ    = BT * (2 * DI) * 4;
  const size_t SZ_XC    = BT * DI * 4;
  const size_t SZ_XC16  = BT * DI * 2;
  const size_t SZ_XDBL  = BT * XPF * 4;
  const size_t SZ_DT16  = BT * DTR * 2;
  const size_t SZ_DELTA = BT * DI * 4;
  const size_t SZ_G16   = BT * DI * 2;
  const size_t SZ_W16I  = (size_t)LAYERS * 2 * DI * DM * 2;
  const size_t SZ_W16X  = (size_t)LAYERS * XPF * DI * 2;
  const size_t SZ_W16D  = (size_t)LAYERS * DI * DTR * 2;

  size_t off = 0;
  float*    h      = (float*)   (ws + off); off += SZ_H;
  _Float16* y16    = (_Float16*)(ws + off); off += SZ_Y16;
  float*    xzb    = (float*)   (ws + off); off += SZ_XZ;
  float*    xcb    = (float*)   (ws + off); off += SZ_XC;
  _Float16* xc16   = (_Float16*)(ws + off); off += SZ_XC16;
  float*    xdbl   = (float*)   (ws + off); off += SZ_XDBL;
  _Float16* dt16   = (_Float16*)(ws + off); off += SZ_DT16;
  float*    delta  = (float*)   (ws + off); off += SZ_DELTA;
  _Float16* g16    = (_Float16*)(ws + off); off += SZ_G16;
  _Float16* w16_in = (_Float16*)(ws + off); off += SZ_W16I;
  _Float16* w16_xp = (_Float16*)(ws + off); off += SZ_W16X;
  _Float16* w16_dt = (_Float16*)(ws + off); off += SZ_W16D;
  _Float16* w16_ow = (_Float16*)(ws + off);

  auto gridFor = [](size_t n) { return dim3((unsigned)((n + 255) / 256)); };

  // ---- convert all weights to f16 once ----
  {
    size_t n;
    n = (size_t)LAYERS * 2 * DI * DM;
    k_f32_to_f16<<<gridFor(n), 256, 0, stream>>>(in_w, w16_in, n);
    n = (size_t)LAYERS * XPF * DI;
    k_f32_to_f16<<<gridFor(n), 256, 0, stream>>>(xp_w, w16_xp, n);
    n = (size_t)LAYERS * DI * DTR;
    k_f32_to_f16<<<gridFor(n), 256, 0, stream>>>(dt_w, w16_dt, n);
    n = (size_t)LAYERS * DM * DI;
    k_f32_to_f16<<<gridFor(n), 256, 0, stream>>>(out_w, w16_ow, n);
  }

  // ---- (B, DM, T) -> (B, T, DM) residual stream ----
  k_transpose_in<<<gridFor(BT * DM), 256, 0, stream>>>(x, h);

  const unsigned gy = (unsigned)(BT / 32);   // M tiles (block tile = 32 rows)

  for (int l = 0; l < LAYERS; ++l) {
    // 1) LayerNorm -> y16
    k_layernorm<<<(unsigned)BT, 256, 0, stream>>>(h, ln_w + l * DM, ln_b + l * DM, y16);

    // 2) in_proj: [BT,DM] x [2DI,DM]^T -> xz [BT,2DI]   (N=2048, full tiles)
    gemm_wmma_f16<false, true><<<dim3((2 * DI) / 256, gy), 256, 0, stream>>>(
        y16, w16_in + (size_t)l * 2 * DI * DM, xzb, (int)BT, 2 * DI, DM);

    // 3) causal conv + SiLU
    k_conv_silu<<<gridFor(BT * DI), 256, 0, stream>>>(
        xzb, conv_w + (size_t)l * DI * DCONV, conv_b + (size_t)l * DI, xcb, xc16);

    // 4) x_proj: [BT,DI] x [XPF,DI]^T -> x_dbl [BT,XPF]  (N=160, ragged)
    gemm_wmma_f16<false, false><<<dim3(1, gy), 256, 0, stream>>>(
        xc16, w16_xp + (size_t)l * XPF * DI, xdbl, (int)BT, XPF, DI);

    // 5) dt_proj: [BT,DTR] x [DI,DTR]^T -> delta_raw [BT,DI]  (N=1024, full)
    k_dt_convert<<<gridFor(BT * DTR), 256, 0, stream>>>(xdbl, dt16);
    gemm_wmma_f16<false, true><<<dim3(DI / 256, gy), 256, 0, stream>>>(
        dt16, w16_dt + (size_t)l * DI * DTR, delta, (int)BT, DI, DTR);
    k_softplus_bias<<<gridFor(BT * DI), 256, 0, stream>>>(delta, dt_b + (size_t)l * DI);

    // 6) selective scan + D skip + SiLU(z) gate -> g16
    k_scan<<<BN * (DI / 64), 64, 0, stream>>>(
        delta, xcb, xdbl, xzb,
        A_log + (size_t)l * DI * DS2, Dp + (size_t)l * DI, g16);

    // 7) out_proj with fused residual: h += [BT,DI] x [DM,DI]^T (N=512, full, ACC)
    gemm_wmma_f16<true, true><<<dim3(DM / 256, gy), 256, 0, stream>>>(
        g16, w16_ow + (size_t)l * DM * DI, h, (int)BT, DM, DI);
  }

  // ---- (B, T, DM) -> (B, DM, T) output ----
  k_transpose_out<<<gridFor(BT * DM), 256, 0, stream>>>(h, (float*)d_out);

  (void)in_sizes; (void)n_in; (void)out_size; (void)ws_size;
}
